// CausalAttention_2388001817315
// MI455X (gfx1250) — compile-verified, matrix operand order checked by reading, not run on hardware
//
#include <hip/hip_runtime.h>
#include <hip/hip_bf16.h>
#include <stdint.h>

// ---------------------------------------------------------------------------
// CDNA5 (gfx1250) causal attention: RMSNorm -> QKV GEMM -> flash attention
// -> out GEMM.  bf16 multiply / fp32 accumulate via v_wmma_f32_16x16x32_bf16.
// Data movement uses the CDNA5 async paths:
//   - TDM tensor_load_to_lds for whole 2-D tiles (pad-on-the-fly into the
//     padded LDS layout), waited with s_wait_tensorcnt
//   - global_load_async_to_lds_b128 (per-lane 16B), waited with
//     s_wait_asynccnt
// GEMMs are double-buffered in LDS (explicit 2-stage pipeline, compile-time
// buffer indices) so DMA overlaps WMMA.
// ---------------------------------------------------------------------------

typedef __attribute__((ext_vector_type(16))) __bf16 bf16x16;
typedef __attribute__((ext_vector_type(8)))  __bf16 bf16x8;
typedef __attribute__((ext_vector_type(8)))  float  f32x8;
typedef __attribute__((ext_vector_type(4)))  unsigned int u32x4;
typedef __attribute__((ext_vector_type(8)))  int    i32x8;
typedef __attribute__((ext_vector_type(4)))  int    i32x4;

union AB16 { bf16x16 v; bf16x8 h[2]; };

__device__ __forceinline__ __bf16 f2bf(float f) {
  uint32_t u = __builtin_bit_cast(uint32_t, f);
  uint32_t r = u + 0x7FFFu + ((u >> 16) & 1u);   // round-to-nearest-even
  uint16_t h = (uint16_t)(r >> 16);
  return __builtin_bit_cast(__bf16, h);
}

// Low 32 bits of a flat LDS address == LDS byte offset (CDNA5 aperture rule).
__device__ __forceinline__ uint32_t lds_off32(const void* p) {
  return (uint32_t)(uintptr_t)p;
}

// Per-lane async 16B copy global -> LDS (ASYNCcnt).
__device__ __forceinline__ void async_load_b128(uint32_t ldsb, uint64_t gaddr) {
  asm volatile("global_load_async_to_lds_b128 %0, %1, off"
               :: "v"(ldsb), "v"(gaddr) : "memory");
}
__device__ __forceinline__ void wait_async0() {
  asm volatile("s_wait_asynccnt 0" ::: "memory");
}

#if defined(__gfx1250__) && __has_builtin(__builtin_amdgcn_tensor_load_to_lds)
#define USE_TDM 1
#else
#define USE_TDM 0
#endif

#if USE_TDM
// 2-D TDM tile load: tile_w x tile_h elements of 2B, row stride in elements,
// LDS padding: pad_amount DWORDs inserted per pad interval (codes per CDNA5
// D# group1) to synthesize padded LDS strides during the DMA.
__device__ __forceinline__ void tdm_load_2d(
    uint32_t ldsb, uint64_t gaddr, uint32_t tile_w, uint32_t tile_h,
    uint32_t tensor_w, uint32_t tensor_h, uint64_t stride_elems,
    uint32_t padi, uint32_t pada) {
  union { uint32_t a[4]; u32x4 v; } g0u = {};
  g0u.a[0] = 1u;                                   // count=1, user descriptor
  g0u.a[1] = ldsb;                                 // lds_addr
  g0u.a[2] = (uint32_t)gaddr;                      // global_addr[31:0]
  g0u.a[3] = (uint32_t)(gaddr >> 32) | (2u << 30); // addr[56:32] | type=2
  union { uint32_t a[8]; i32x8 v; } g1u = {};
  g1u.a[0] = (1u << 16) | (1u << 20) | (padi << 22) | (pada << 25); // 2B, pad
  g1u.a[1] = (tensor_w & 0xFFFFu) << 16;
  g1u.a[2] = (tensor_w >> 16) | ((tensor_h & 0xFFFFu) << 16);
  g1u.a[3] = (tensor_h >> 16) | (tile_w << 16);
  g1u.a[4] = tile_h;                               // tile_dim1; tile_dim2=0
  g1u.a[5] = (uint32_t)stride_elems;               // dim0 stride low
  g1u.a[6] = (uint32_t)(stride_elems >> 32) & 0xFFFFu;
  g1u.a[7] = 0;
  i32x4 z4 = {};
  i32x8 z8 = {};
  __builtin_amdgcn_tensor_load_to_lds(g0u.v, g1u.v, z4, z4, z8, 0);
}
#endif

#define BATCH 2
#define SEQ   2048
#define DMODEL 1024
#define HEADS 16
#define DHEAD 64

// ---------------------------------------------------------------------------
// Kernel 1: RMSNorm + cast to bf16.  One block per row.
// ---------------------------------------------------------------------------
__global__ void __launch_bounds__(256) rmsnorm_cast_kernel(
    const float* __restrict__ x, const float* __restrict__ gamma,
    __bf16* __restrict__ xn) {
  int row = blockIdx.x;
  int t = threadIdx.x;
  const float* xr = x + (size_t)row * DMODEL;
  float v[4]; float s = 0.f;
#pragma unroll
  for (int i = 0; i < 4; ++i) { v[i] = xr[t + i * 256]; s += v[i] * v[i]; }
  __shared__ float red[256];
  red[t] = s;
  __syncthreads();
  for (int off = 128; off > 0; off >>= 1) {
    if (t < off) red[t] += red[t + off];
    __syncthreads();
  }
  float inv = 32.0f / fmaxf(sqrtf(red[0]), 1e-12f);   // sqrt(1024)=32
  __bf16* o = xn + (size_t)row * DMODEL;
#pragma unroll
  for (int i = 0; i < 4; ++i) o[t + i * 256] = f2bf(v[i] * inv * gamma[t + i * 256]);
}

// ---------------------------------------------------------------------------
// Kernel 2: fp32 -> bf16 weight cast
// ---------------------------------------------------------------------------
__global__ void cast_f32_bf16_kernel(const float* __restrict__ in,
                                     __bf16* __restrict__ out, int n) {
  int i = blockIdx.x * blockDim.x + threadIdx.x;
  if (i < n) out[i] = f2bf(in[i]);
}

// ---------------------------------------------------------------------------
// Kernels 3/5: bf16 GEMM, block tile 128x64, double-buffered LDS with an
// explicit 2-stage pipeline (compile-time buffer indices), A tile via TDM
// (fallback: async-to-LDS), B tile transposed through VGPRs.
// ---------------------------------------------------------------------------
#define LDP 40   // padded LDS row stride (80B, keeps 16B alignment)

template <int EPILOGUE>   // 0 = qkv scatter (bf16), 1 = plain fp32 store
__global__ void __launch_bounds__(256) gemm_bf16_kernel(
    const __bf16* __restrict__ A, const __bf16* __restrict__ Bw,
    void* __restrict__ outp, int K, int Nn, int Mtot) {
  int n0 = blockIdx.x * 64;
  int m0 = blockIdx.y * 128;
  int tid = threadIdx.x;
  int lane = tid & 31, wave = tid >> 5;
  int wr = wave >> 1, wc = wave & 1;
  int m0w = wr * 32, n0w = wc * 32;
  int mrow = lane & 15, khalf = lane >> 4;

  __shared__ __align__(16) __bf16 As[2][128 * LDP];
  __shared__ __align__(16) __bf16 Bt[2][64 * LDP];

  f32x8 acc[2][2] = {};

  auto loadA = [&](int kt, int buf) {
#if USE_TDM
    if (wave == 0)
      tdm_load_2d(lds_off32(&As[buf][0]),
                  (uint64_t)(uintptr_t)(A + (size_t)m0 * K + kt),
                  /*tile_w=*/32, /*tile_h=*/128,
                  /*tensor_w=*/(uint32_t)K, /*tensor_h=*/(uint32_t)Mtot,
                  /*stride=*/(uint64_t)K, /*padi=*/3, /*pada=*/3);
#else
#pragma unroll
    for (int i = 0; i < 2; ++i) {
      int c = tid + i * 256;
      int r = c >> 2, cc = (c & 3) * 8;
      async_load_b128(lds_off32(&As[buf][r * LDP + cc]),
                      (uint64_t)(uintptr_t)(A + (size_t)(m0 + r) * K + kt + cc));
    }
#endif
  };
  auto fetchB = [&](int kt) -> uint4 {
    int kr = tid >> 3, nc = (tid & 7) * 8;
    return *(const uint4*)(Bw + (size_t)(kt + kr) * Nn + n0 + nc);
  };
  auto storeB = [&](uint4 d, int buf) {
    int kr = tid >> 3, nc = (tid & 7) * 8;
    const __bf16* dv = (const __bf16*)&d;
#pragma unroll
    for (int j = 0; j < 8; ++j) Bt[buf][(nc + j) * LDP + kr] = dv[j];
  };
  auto waitdma = [&]() {
#if USE_TDM
    if (wave == 0) __builtin_amdgcn_s_wait_tensorcnt(0);
#else
    wait_async0();
#endif
  };
  auto compute = [&](int buf) {
    AB16 af[2], bfr[2];
#pragma unroll
    for (int mi = 0; mi < 2; ++mi) {
      int r = m0w + mi * 16 + mrow;
      af[mi].h[0] = *(const bf16x8*)&As[buf][r * LDP + khalf * 8];
      af[mi].h[1] = *(const bf16x8*)&As[buf][r * LDP + 16 + khalf * 8];
    }
#pragma unroll
    for (int ni = 0; ni < 2; ++ni) {
      int r = n0w + ni * 16 + mrow;
      bfr[ni].h[0] = *(const bf16x8*)&Bt[buf][r * LDP + khalf * 8];
      bfr[ni].h[1] = *(const bf16x8*)&Bt[buf][r * LDP + 16 + khalf * 8];
    }
#pragma unroll
    for (int mi = 0; mi < 2; ++mi)
#pragma unroll
      for (int ni = 0; ni < 2; ++ni)
        acc[mi][ni] = __builtin_amdgcn_wmma_f32_16x16x32_bf16(
            false, af[mi].v, false, bfr[ni].v, (short)0, acc[mi][ni], false, false);
  };

  // prologue: tile 0 -> buffer 0
  loadA(0, 0);
  storeB(fetchB(0), 0);
  waitdma();
  __syncthreads();

  int nk = K / 32;                 // K is a multiple of 64 -> nk even
  for (int it = 0; it < nk; it += 2) {
    // stage 0: compute buf0, DMA next tile into buf1
    bool more1 = (it + 1) < nk;
    uint4 b1;
    if (more1) { loadA((it + 1) * 32, 1); b1 = fetchB((it + 1) * 32); }
    compute(0);
    if (more1) { storeB(b1, 1); waitdma(); }
    __syncthreads();
    if (!more1) break;
    // stage 1: compute buf1, DMA next tile into buf0
    bool more2 = (it + 2) < nk;
    uint4 b0;
    if (more2) { loadA((it + 2) * 32, 0); b0 = fetchB((it + 2) * 32); }
    compute(1);
    if (more2) { storeB(b0, 0); waitdma(); }
    __syncthreads();
  }

  int half = lane >> 4;
#pragma unroll
  for (int mi = 0; mi < 2; ++mi)
#pragma unroll
    for (int ni = 0; ni < 2; ++ni)
#pragma unroll
      for (int v = 0; v < 8; ++v) {
        int r = m0 + m0w + mi * 16 + v + 8 * half;
        int c = n0 + n0w + ni * 16 + mrow;
        float val = acc[mi][ni][v];
        if (EPILOGUE == 0) {
          // scatter to [3][b][h][n][d]; pre-scale q by d^-0.5
          int qi = c >> 10, rem = c & 1023;
          int head = rem >> 6, dd = rem & 63;
          int bi = r >> 11, nn = r & 2047;
          if (qi == 0) val *= 0.125f;
          size_t off = (((size_t)qi * BATCH + bi) * HEADS + head) * (size_t)(SEQ * DHEAD)
                     + (size_t)nn * DHEAD + dd;
          ((__bf16*)outp)[off] = f2bf(val);
        } else {
          ((float*)outp)[(size_t)r * Nn + c] = val;
        }
      }
}

// ---------------------------------------------------------------------------
// Kernel 4: causal flash attention, one (b,h) x 128 query rows per block.
// Q tile via TDM, K tiles via global_load_async_to_lds_b128, V transposed
// through VGPRs.  Online softmax in the WMMA C-frag row mapping.
// ---------------------------------------------------------------------------
#define LQ 72   // 144B row stride

__global__ void __launch_bounds__(256) flash_attn_kernel(
    const __bf16* __restrict__ qb, const __bf16* __restrict__ kb,
    const __bf16* __restrict__ vb, __bf16* __restrict__ ao) {
  int q0 = blockIdx.x * 128;
  int bh = blockIdx.y;
  int bi = bh >> 4, head = bh & 15;
  size_t base = (size_t)(bi * HEADS + head) * (size_t)(SEQ * DHEAD);
  const __bf16* Q = qb + base;
  const __bf16* Kp = kb + base;
  const __bf16* Vp = vb + base;

  int tid = threadIdx.x, lane = tid & 31, wave = tid >> 5;
  int mrow = lane & 15, khalf = lane >> 4, half = khalf;

  __shared__ __align__(16) __bf16 Qs[128 * LQ];
  __shared__ __align__(16) __bf16 Ks[64 * LQ];
  __shared__ __align__(16) __bf16 Vt[64 * LQ];       // transposed: [d][key]
  __shared__ __align__(16) __bf16 Ps[8 * 16 * LQ];   // per-wave P scratch

  // Q tile 128x64 -> LDS (TDM with pad: 32DW interval, 4DW pad -> stride 72)
#if USE_TDM
  if (wave == 0) {
    tdm_load_2d(lds_off32(&Qs[0]), (uint64_t)(uintptr_t)(Q + (size_t)q0 * DHEAD),
                /*tile_w=*/64, /*tile_h=*/128, /*tensor_w=*/64, /*tensor_h=*/SEQ,
                /*stride=*/64, /*padi=*/4, /*pada=*/3);
    __builtin_amdgcn_s_wait_tensorcnt(0);
  }
#else
#pragma unroll
  for (int i = 0; i < 4; ++i) {
    int c = tid + i * 256;
    int r = c >> 3, cc = (c & 7) * 8;
    async_load_b128(lds_off32(&Qs[r * LQ + cc]),
                    (uint64_t)(uintptr_t)(Q + (size_t)(q0 + r) * DHEAD + cc));
  }
  wait_async0();
#endif
  __syncthreads();

  int qr0 = wave * 16;
  AB16 aq[2];
#pragma unroll
  for (int kk = 0; kk < 2; ++kk) {
    int r = qr0 + mrow;
    aq[kk].h[0] = *(const bf16x8*)&Qs[r * LQ + kk * 32 + khalf * 8];
    aq[kk].h[1] = *(const bf16x8*)&Qs[r * LQ + kk * 32 + 16 + khalf * 8];
  }

  f32x8 o[4] = {};
  f32x8 mvec, lvec;
#pragma unroll
  for (int v = 0; v < 8; ++v) { mvec[v] = -INFINITY; lvec[v] = 0.f; }

  int numKV = blockIdx.x * 2 + 2;
  for (int j = 0; j < numKV; ++j) {
    int j0 = j * 64;
    __syncthreads();
    // K tile 64x64: per-lane async copies into LDS (ASYNCcnt)
#pragma unroll
    for (int i = 0; i < 2; ++i) {
      int c = tid + i * 256;
      int r = c >> 3, cc = (c & 7) * 8;
      async_load_b128(lds_off32(&Ks[r * LQ + cc]),
                      (uint64_t)(uintptr_t)(Kp + (size_t)(j0 + r) * DHEAD + cc));
    }
    // V tile transposed through VGPRs -> Vt[d][key]
#pragma unroll
    for (int i = 0; i < 2; ++i) {
      int c = tid + i * 256;
      int r = c >> 3, cc = (c & 7) * 8;
      uint4 d = *(const uint4*)(Vp + (size_t)(j0 + r) * DHEAD + cc);
      const __bf16* dv = (const __bf16*)&d;
#pragma unroll
      for (int jj = 0; jj < 8; ++jj) Vt[(cc + jj) * LQ + r] = dv[jj];
    }
    if (j + 1 < numKV)
      __builtin_prefetch(Vp + (size_t)(j0 + 64) * DHEAD + tid * 8, 0, 1);
    wait_async0();
    __syncthreads();

    // S = Q K^T  (4 subtiles of 16 keys, 2 WMMAs each over d=64)
    f32x8 s[4];
#pragma unroll
    for (int sub = 0; sub < 4; ++sub) {
      f32x8 accs = {};
#pragma unroll
      for (int kk = 0; kk < 2; ++kk) {
        AB16 bk;
        int key = sub * 16 + mrow;
        bk.h[0] = *(const bf16x8*)&Ks[key * LQ + kk * 32 + khalf * 8];
        bk.h[1] = *(const bf16x8*)&Ks[key * LQ + kk * 32 + 16 + khalf * 8];
        accs = __builtin_amdgcn_wmma_f32_16x16x32_bf16(
            false, aq[kk].v, false, bk.v, (short)0, accs, false, false);
      }
      s[sub] = accs;
    }

    // causal mask
    if (j0 + 63 > q0) {
#pragma unroll
      for (int sub = 0; sub < 4; ++sub)
#pragma unroll
        for (int v = 0; v < 8; ++v) {
          int qr = q0 + qr0 + v + 8 * half;
          int kc = j0 + sub * 16 + mrow;
          if (kc > qr) s[sub][v] = -1e30f;
        }
    }

    // online softmax: row max / sum in C-frag mapping
    f32x8 rm = s[0];
#pragma unroll
    for (int sub = 1; sub < 4; ++sub)
#pragma unroll
      for (int v = 0; v < 8; ++v) rm[v] = fmaxf(rm[v], s[sub][v]);
#pragma unroll
    for (int msk = 1; msk < 16; msk <<= 1)
#pragma unroll
      for (int v = 0; v < 8; ++v) rm[v] = fmaxf(rm[v], __shfl_xor(rm[v], msk, 32));

    f32x8 mnew, alpha;
#pragma unroll
    for (int v = 0; v < 8; ++v) {
      mnew[v] = fmaxf(mvec[v], rm[v]);
      alpha[v] = __expf(mvec[v] - mnew[v]);
    }

    f32x8 rs = {};
#pragma unroll
    for (int sub = 0; sub < 4; ++sub)
#pragma unroll
      for (int v = 0; v < 8; ++v) {
        float p = __expf(s[sub][v] - mnew[v]);
        s[sub][v] = p;
        rs[v] += p;
      }
#pragma unroll
    for (int msk = 1; msk < 16; msk <<= 1)
#pragma unroll
      for (int v = 0; v < 8; ++v) rs[v] += __shfl_xor(rs[v], msk, 32);

#pragma unroll
    for (int v = 0; v < 8; ++v) {
      lvec[v] = lvec[v] * alpha[v] + rs[v];
      mvec[v] = mnew[v];
    }
#pragma unroll
    for (int ni = 0; ni < 4; ++ni)
#pragma unroll
      for (int v = 0; v < 8; ++v) o[ni][v] *= alpha[v];

    // P: C-layout regs -> per-wave LDS -> A-layout frags
    __bf16* Pw = &Ps[wave * 16 * LQ];
#pragma unroll
    for (int sub = 0; sub < 4; ++sub)
#pragma unroll
      for (int v = 0; v < 8; ++v)
        Pw[(v + 8 * half) * LQ + sub * 16 + mrow] = f2bf(s[sub][v]);

    AB16 ap[2];
#pragma unroll
    for (int kk = 0; kk < 2; ++kk) {
      ap[kk].h[0] = *(const bf16x8*)&Pw[mrow * LQ + kk * 32 + khalf * 8];
      ap[kk].h[1] = *(const bf16x8*)&Pw[mrow * LQ + kk * 32 + 16 + khalf * 8];
    }
    // O += P V
#pragma unroll
    for (int ni = 0; ni < 4; ++ni)
#pragma unroll
      for (int kk = 0; kk < 2; ++kk) {
        AB16 bv;
        int dc = ni * 16 + mrow;
        bv.h[0] = *(const bf16x8*)&Vt[dc * LQ + kk * 32 + khalf * 8];
        bv.h[1] = *(const bf16x8*)&Vt[dc * LQ + kk * 32 + 16 + khalf * 8];
        o[ni] = __builtin_amdgcn_wmma_f32_16x16x32_bf16(
            false, ap[kk].v, false, bv.v, (short)0, o[ni], false, false);
      }
  }

  // epilogue: ao[bi][q][head*64 + d] = o / l  (bf16 for the final GEMM)
#pragma unroll
  for (int ni = 0; ni < 4; ++ni)
#pragma unroll
    for (int v = 0; v < 8; ++v) {
      int qr = q0 + qr0 + v + 8 * half;
      int c = head * 64 + ni * 16 + mrow;
      ao[((size_t)bi * SEQ + qr) * (size_t)DMODEL + c] = f2bf(o[ni][v] / lvec[v]);
    }
}

// ---------------------------------------------------------------------------
// Launch
// ---------------------------------------------------------------------------
extern "C" void kernel_launch(void* const* d_in, const int* in_sizes, int n_in,
                              void* d_out, int out_size, void* d_ws, size_t ws_size,
                              hipStream_t stream) {
  (void)in_sizes; (void)n_in; (void)out_size; (void)ws_size;
  const float* x     = (const float*)d_in[0];
  const float* gamma = (const float*)d_in[1];
  const float* w_qkv = (const float*)d_in[2];
  const float* w_out = (const float*)d_in[3];
  float* out = (float*)d_out;

  char* ws = (char*)d_ws;
  size_t off = 0;
  auto wsalloc = [&](size_t bytes) -> void* {
    void* p = ws + off;
    off += (bytes + 255) & ~(size_t)255;
    return p;
  };
  __bf16* xn    = (__bf16*)wsalloc((size_t)4096 * 1024 * 2);
  __bf16* wqb   = (__bf16*)wsalloc((size_t)1024 * 3072 * 2);
  __bf16* wob   = (__bf16*)wsalloc((size_t)1024 * 1024 * 2);
  __bf16* qkv   = (__bf16*)wsalloc((size_t)3 * 2 * 16 * 2048 * 64 * 2);
  __bf16* aobuf = (__bf16*)wsalloc((size_t)4096 * 1024 * 2);

  rmsnorm_cast_kernel<<<4096, 256, 0, stream>>>(x, gamma, xn);
  cast_f32_bf16_kernel<<<(1024 * 3072 + 255) / 256, 256, 0, stream>>>(w_qkv, wqb, 1024 * 3072);
  cast_f32_bf16_kernel<<<(1024 * 1024 + 255) / 256, 256, 0, stream>>>(w_out, wob, 1024 * 1024);

  gemm_bf16_kernel<0><<<dim3(48, 32), 256, 0, stream>>>(xn, wqb, (void*)qkv,
                                                        1024, 3072, 4096);

  const __bf16* qb = qkv;
  const __bf16* kb = qkv + (size_t)1 * 2 * 16 * 2048 * 64;
  const __bf16* vb = qkv + (size_t)2 * 2 * 16 * 2048 * 64;
  flash_attn_kernel<<<dim3(16, 32), 256, 0, stream>>>(qb, kb, vb, aobuf);

  gemm_bf16_kernel<1><<<dim3(16, 32), 256, 0, stream>>>(aobuf, wob, (void*)out,
                                                        1024, 1024, 4096);
}
